// NodeModelIn_43843026158106
// MI455X (gfx1250) — compile-verified, matrix operand order checked by reading, not run on hardware
//
#include <hip/hip_runtime.h>
#include <hip/hip_fp16.h>
#include <float.h>
#include <stdint.h>

typedef __attribute__((ext_vector_type(16))) _Float16 v16h;
typedef __attribute__((ext_vector_type(8)))  _Float16 v8h;
typedef __attribute__((ext_vector_type(8)))  float    v8f;

#define EDGE_OUT 64
#define HID      256
#define NODE_OUT 128
#define FAN_IN   193
#define K_PAD    224           // 193 padded to 7 * 32 (K-steps of 16x16x32 WMMA)
#define KSTEPS1  (K_PAD / 32)  // 7
#define KSTEPS2  (HID / 32)    // 8
#define NT1      (HID / 16)    // 16 column tiles in GEMM1
#define NT2      (NODE_OUT / 16) // 8 column tiles in GEMM2

// ---------------------------------------------------------------------------
// float atomic max via signed/unsigned integer trick (monotone bit encodings)
// ---------------------------------------------------------------------------
__device__ __forceinline__ void atomicMaxF32(float* addr, float v) {
    if (v >= 0.0f) atomicMax((int*)addr, __float_as_int(v));
    else           atomicMin((unsigned int*)addr, __float_as_uint(v));
}

// ---------------------------------------------------------------------------
// Kernel 1: initialize scatter accumulators (every launch: scatter accumulates)
// ---------------------------------------------------------------------------
__global__ void init_ws_kernel(float* __restrict__ out1, float* __restrict__ out2,
                               float* __restrict__ cnt, int n_nodes) {
    int i = blockIdx.x * blockDim.x + threadIdx.x;
    int tot = n_nodes * EDGE_OUT;
    if (i < tot) {
        out1[i] = 0.0f;
        out2[i] = -FLT_MAX;
    }
    if (i < n_nodes) cnt[i] = 0.0f;
}

// ---------------------------------------------------------------------------
// Kernel 2: convert weights to f16 AND pre-swizzle into WMMA B-fragment order.
// Layout: frag[(ks * NTILES + t) * 32 + lane][i]  (16 contiguous halves/lane)
//   per ISA 7.12.2 16-bit fragment layout:
//     kb = (lane>>4)*8 ; element i -> K = ks*32 + kb + (i<8 ? i : i+8)
//     N = t*16 + (lane&15)
// So the MLP kernel loads one 32-byte contiguous v16h per lane (coalesced).
// ---------------------------------------------------------------------------
__global__ void convert_weights_kernel(const float* __restrict__ W1,
                                       const float* __restrict__ W2,
                                       _Float16* __restrict__ W1p,
                                       _Float16* __restrict__ W2p) {
    int j = blockIdx.x * blockDim.x + threadIdx.x;
    // ---- W1 fragments: KSTEPS1 * NT1 * 32 lanes * 16 halves = 57344 ----
    if (j < KSTEPS1 * NT1 * 32 * 16) {
        int i    = j & 15;
        int lane = (j >> 4) & 31;
        int t    = (j >> 9) % NT1;
        int ks   = (j >> 9) / NT1;
        int kb   = (lane >> 4) * 8;
        int k    = ks * 32 + kb + ((i < 8) ? i : (i + 8));
        int n    = t * 16 + (lane & 15);
        W1p[j] = (k < FAN_IN) ? (_Float16)W1[k * HID + n] : (_Float16)0.0f;
    }
    // ---- W2 fragments: KSTEPS2 * NT2 * 32 * 16 = 32768 ----
    if (j < KSTEPS2 * NT2 * 32 * 16) {
        int i    = j & 15;
        int lane = (j >> 4) & 31;
        int t    = (j >> 9) % NT2;
        int ks   = (j >> 9) / NT2;
        int kb   = (lane >> 4) * 8;
        int k    = ks * 32 + kb + ((i < 8) ? i : (i + 8));
        int n    = t * 16 + (lane & 15);
        W2p[j] = (_Float16)W2[k * NODE_OUT + n];
    }
}

// ---------------------------------------------------------------------------
// Kernel 3: one-pass edge scatter. 16 threads/edge, float4 loads. Targets
// (~52MB) are L2-resident on MI455X (192MB L2): atomics resolve in L2.
// ---------------------------------------------------------------------------
__global__ void edge_scatter_kernel(const float* __restrict__ edge_attr,
                                    const int* __restrict__ col,
                                    float* __restrict__ out1,
                                    float* __restrict__ out2,
                                    float* __restrict__ cnt,
                                    long n_edges) {
    long t = (long)blockIdx.x * blockDim.x + threadIdx.x;
    long e = t >> 4;
    int fg = (int)(t & 15);
    if (e >= n_edges) return;
    int node = col[e];
    const float4 v = ((const float4*)(edge_attr + e * EDGE_OUT))[fg];
    float* o1 = out1 + (long)node * EDGE_OUT + fg * 4;
    float* o2 = out2 + (long)node * EDGE_OUT + fg * 4;
    atomicAdd(o1 + 0, v.x);
    atomicAdd(o1 + 1, v.y);
    atomicAdd(o1 + 2, v.z);
    atomicAdd(o1 + 3, v.w);
    atomicMaxF32(o2 + 0, v.x);
    atomicMaxF32(o2 + 1, v.y);
    atomicMaxF32(o2 + 2, v.z);
    atomicMaxF32(o2 + 3, v.w);
    if (fg == 0) atomicAdd(cnt + node, 1.0f);
}

// ---------------------------------------------------------------------------
// A-fragment loader: two contiguous 16-byte (v8h) LDS loads -> ds_load_b128,
// merged into a v16h with a zero-cost shuffle.
// ---------------------------------------------------------------------------
__device__ __forceinline__ v16h load_a_frag(const _Float16* row, int koff) {
    v8h lo = *(const v8h*)(row + koff);
    v8h hi = *(const v8h*)(row + koff + 16);
    return __builtin_shufflevector(lo, hi, 0, 1, 2, 3, 4, 5, 6, 7,
                                   8, 9, 10, 11, 12, 13, 14, 15);
}

// ---------------------------------------------------------------------------
// Kernel 4: fused feat-build + 2-layer MLP. Block = 128 threads (4 wave32),
// one 16-node tile per block.
// ---------------------------------------------------------------------------
__global__ __launch_bounds__(128)
void mlp_wmma_kernel(const float* __restrict__ out1, const float* __restrict__ out2,
                     const float* __restrict__ cnt, const float* __restrict__ u,
                     const int* __restrict__ batch,
                     const _Float16* __restrict__ W1p, const float* __restrict__ b1,
                     const _Float16* __restrict__ W2p, const float* __restrict__ b2,
                     float* __restrict__ out, int n_nodes) {
    __shared__ alignas(16) _Float16 featL[16][K_PAD];  // 7 KB
    __shared__ alignas(16) _Float16 hL[16][HID];       // 8 KB

    const int tid  = threadIdx.x;
    const int base = blockIdx.x * 16;

    // -------- Stage 0: build feat tile in LDS (f16), 4 k-values/iter --------
    for (int idx = tid; idx < 16 * (K_PAD / 4); idx += 128) {
        int row = idx / (K_PAD / 4);
        int k4  = idx % (K_PAD / 4);      // group of 4 consecutive k
        int node = base + row;
        float4 v = make_float4(0.f, 0.f, 0.f, 0.f);
        if (node < n_nodes) {
            float c = cnt[node];
            if (k4 < 16) {                                   // scatter_add
                v = ((const float4*)(out1 + (long)node * EDGE_OUT))[k4];
            } else if (k4 < 32) {                            // scatter_max
                if (c > 0.0f)
                    v = ((const float4*)(out2 + (long)node * EDGE_OUT))[k4 - 16];
            } else if (k4 < 48) {                            // scatter_mean
                float inv = 1.0f / fmaxf(c, 1.0f);
                float4 s = ((const float4*)(out1 + (long)node * EDGE_OUT))[k4 - 32];
                v = make_float4(s.x * inv, s.y * inv, s.z * inv, s.w * inv);
            } else if (k4 == 48) {                           // u[batch] + pad
                v.x = u[batch[node]];
            } // k4 in (48,56): zero padding
        }
        _Float16* dst = &featL[row][k4 * 4];
        dst[0] = (_Float16)v.x;
        dst[1] = (_Float16)v.y;
        dst[2] = (_Float16)v.z;
        dst[3] = (_Float16)v.w;
    }
    __syncthreads();

    const int lane = tid & 31;
    const int wave = tid >> 5;
    const int m    = lane & 15;
    const int kb   = (lane >> 4) * 8;

    // -------- Stage 1: h = gelu(feat @ W1 + b1) --------
    // wave w owns HID column tiles {4w .. 4w+3} (64 columns)
    v8f acc[4] = {};
    for (int ks = 0; ks < KSTEPS1; ++ks) {
        v16h a = load_a_frag(&featL[m][0], ks * 32 + kb);
        #pragma unroll
        for (int t = 0; t < 4; ++t) {
            int tg = wave * 4 + t;
            v16h b = *(const v16h*)(W1p + ((size_t)(ks * NT1 + tg) * 32 + lane) * 16);
            acc[t] = __builtin_amdgcn_wmma_f32_16x16x32_f16(
                false, a, false, b, (short)0, acc[t], false, false);
        }
    }
    // epilogue: bias + exact GELU -> LDS (f16)
    #pragma unroll
    for (int t = 0; t < 4; ++t) {
        int ncol = (wave * 4 + t) * 16 + (lane & 15);
        float bias = b1[ncol];
        #pragma unroll
        for (int r = 0; r < 8; ++r) {
            int mm = r + (lane >> 4) * 8;      // C/D layout: VGPR r -> row M
            float xv = acc[t][r] + bias;
            float g  = 0.5f * xv * (1.0f + erff(xv * 0.70710678118654752f));
            hL[mm][ncol] = (_Float16)g;
        }
    }
    __syncthreads();

    // -------- Stage 2: out = h @ W2 + b2 --------
    // wave w owns output column tiles {2w, 2w+1} (32 columns)
    v8f acc2[2] = {};
    for (int ks = 0; ks < KSTEPS2; ++ks) {
        v16h a = load_a_frag(&hL[m][0], ks * 32 + kb);
        #pragma unroll
        for (int t = 0; t < 2; ++t) {
            int tg = wave * 2 + t;
            v16h b = *(const v16h*)(W2p + ((size_t)(ks * NT2 + tg) * 32 + lane) * 16);
            acc2[t] = __builtin_amdgcn_wmma_f32_16x16x32_f16(
                false, a, false, b, (short)0, acc2[t], false, false);
        }
    }
    #pragma unroll
    for (int t = 0; t < 2; ++t) {
        int ncol = (wave * 2 + t) * 16 + (lane & 15);
        float bias = b2[ncol];
        #pragma unroll
        for (int r = 0; r < 8; ++r) {
            int node = base + r + (lane >> 4) * 8;
            if (node < n_nodes) {
                out[(long)node * NODE_OUT + ncol] = acc2[t][r] + bias;
            }
        }
    }
}

// ---------------------------------------------------------------------------
// kernel_launch
//   inputs: 0:x[N,7] 1:edge_index[2,E] 2:edge_attr[E,64] 3:u[16,1]
//           4:batch[N] 5:W1[193,256] 6:b1[256] 7:W2[256,128] 8:b2[128]
// ---------------------------------------------------------------------------
extern "C" void kernel_launch(void* const* d_in, const int* in_sizes, int n_in,
                              void* d_out, int out_size, void* d_ws, size_t ws_size,
                              hipStream_t stream) {
    const int*   edge_index = (const int*)d_in[1];
    const float* edge_attr  = (const float*)d_in[2];
    const float* u          = (const float*)d_in[3];
    const int*   batch      = (const int*)d_in[4];
    const float* W1         = (const float*)d_in[5];
    const float* b1         = (const float*)d_in[6];
    const float* W2         = (const float*)d_in[7];
    const float* b2         = (const float*)d_in[8];

    const int  n_nodes = in_sizes[0] / 7;
    const long n_edges = (long)in_sizes[1] / 2;
    const int* col     = edge_index + n_edges;   // edge_index[1, :]

    // workspace layout
    char* ws = (char*)d_ws;
    float* out1 = (float*)ws;  ws += (size_t)n_nodes * EDGE_OUT * sizeof(float);
    float* out2 = (float*)ws;  ws += (size_t)n_nodes * EDGE_OUT * sizeof(float);
    float* cntp = (float*)ws;  ws += (size_t)n_nodes * sizeof(float);
    uintptr_t p = (uintptr_t)ws; p = (p + 255) & ~(uintptr_t)255; ws = (char*)p;
    _Float16* W1p = (_Float16*)ws; ws += (size_t)K_PAD * HID * sizeof(_Float16);
    _Float16* W2p = (_Float16*)ws; ws += (size_t)HID * NODE_OUT * sizeof(_Float16);

    // 1) init accumulators
    {
        int tot = n_nodes * EDGE_OUT;
        int blocks = (tot + 255) / 256;
        init_ws_kernel<<<blocks, 256, 0, stream>>>(out1, out2, cntp, n_nodes);
    }
    // 2) weight conversion + fragment pre-swizzle
    {
        int tot = KSTEPS1 * NT1 * 32 * 16;   // 57344 >= 32768
        int blocks = (tot + 255) / 256;
        convert_weights_kernel<<<blocks, 256, 0, stream>>>(W1, W2, W1p, W2p);
    }
    // 3) edge scatter (L2-resident atomics)
    {
        long threads = n_edges * 16;
        long blocks = (threads + 255) / 256;
        edge_scatter_kernel<<<(unsigned)blocks, 256, 0, stream>>>(
            edge_attr, col, out1, out2, cntp, n_edges);
    }
    // 4) fused feat + MLP via WMMA
    {
        int ntiles = (n_nodes + 15) / 16;
        mlp_wmma_kernel<<<ntiles, 128, 0, stream>>>(
            out1, out2, cntp, u, batch, W1p, b1, W2p, b2, (float*)d_out, n_nodes);
    }
}
